// GraphNetwork_83468394431127
// MI455X (gfx1250) — matrix operand order repaired and checked by reading, not compiled, not run on hardware
//
#include <hip/hip_runtime.h>

typedef __attribute__((ext_vector_type(16))) __bf16 v16bf;
typedef __attribute__((ext_vector_type(8)))  float  v8f;

#define GN_F     96
#define GN_NT    6      // 96/16 output tiles
#define GN_WAVES 4      // waves (tiles) per block

__device__ __forceinline__ unsigned short f32_to_bf16(float f) {
  union { float f; unsigned int u; } v; v.f = f;
  unsigned int u = v.u;
  u += 0x7FFFu + ((u >> 16) & 1u);          // round-to-nearest-even
  return (unsigned short)(u >> 16);
}

union BFV { v16bf v; unsigned short u[16]; };

// ---------------------------------------------------------------------------
// Pre-pack f32 weights [S, K, 96] into bf16 WMMA B-fragments:
// layout [S][K/32][6 ntiles][32 lanes][16 vals], so each lane's fragment is a
// contiguous 32-byte v16bf load. B layout (16-bit, 32x16): lanes 0-15 hold
// K=0..7,16..23 (col = lane), lanes 16-31 hold K=8..15,24..31 (col = lane-16).
// ---------------------------------------------------------------------------
__global__ void pack_b_kernel(const float* __restrict__ W, unsigned short* __restrict__ out,
                              int K, int total) {
  int tid = blockIdx.x * blockDim.x + threadIdx.x;
  if (tid >= total) return;
  int j    = tid & 15;
  int lane = (tid >> 4) & 31;
  int rem  = tid >> 9;
  int nt   = rem % GN_NT; rem /= GN_NT;
  int KT   = K >> 5;
  int kt   = rem % KT;
  int s    = rem / KT;
  int k    = (kt << 5) + ((lane >> 4) << 3) + j + ((j >> 3) << 3);
  int c    = (nt << 4) + (lane & 15);
  out[tid] = f32_to_bf16(W[((size_t)s * K + k) * GN_F + c]);
}

__global__ void zero_kernel(float* __restrict__ p, long long n) {
  long long i = (long long)blockIdx.x * blockDim.x + threadIdx.x;
  if (i < n) p[i] = 0.0f;
}

// Embedder: out[r][f] = b[f] + sum_k X[r][k] * W[k][f]   (K is 8/16/32: tiny)
__global__ void embed_kernel(const float* __restrict__ X, const float* __restrict__ W,
                             const float* __restrict__ b, float* __restrict__ out,
                             long long rows, int K) {
  long long tid = (long long)blockIdx.x * blockDim.x + threadIdx.x;
  if (tid >= rows * GN_F) return;
  long long r = tid / GN_F;
  int f = (int)(tid % GN_F);
  float a = b[f];
  const float* x = X + r * K;
  for (int k = 0; k < K; ++k) a += x[k] * W[(size_t)k * GN_F + f];
  out[tid] = a;
}

// ---------------------------------------------------------------------------
// Fused edge MLP step: x = [e | n[snd] | n[rcv] | g]  (16 edges x 384, bf16 LDS)
// h = leaky(x @ W1 + b1) ; e' = h @ W2 + b2 ; in-place e write + atomic
// scatter into sent/recv segment sums and esum.
// ---------------------------------------------------------------------------
__global__ __launch_bounds__(32 * GN_WAVES) void edge_update_kernel(
    float* __restrict__ e, const float* __restrict__ n, const float* __restrict__ g,
    const int* __restrict__ snd, const int* __restrict__ rcv,
    const unsigned short* __restrict__ W1p, const float* __restrict__ b1,
    const unsigned short* __restrict__ W2p, const float* __restrict__ b2,
    float* __restrict__ sent, float* __restrict__ recvb, float* __restrict__ esum,
    int ntiles, int E)
{
  __shared__ unsigned short lx[GN_WAVES][16][4 * GN_F];
  const int lane = threadIdx.x & 31;
  const int w    = threadIdx.x >> 5;
  const int tile = blockIdx.x * GN_WAVES + w;
  const bool active = tile < ntiles;
  const int gq  = lane >> 4;
  const int col = lane & 15;
  const int m16 = lane & 15;
  const int e0  = tile * 16;

  if (active) {
    for (int m = 0; m < 16; ++m) {
      const int edge = e0 + m;
      if (edge >= E) break;
      const float* er = e + (size_t)edge * GN_F;
      const float* ns = n + (size_t)snd[edge] * GN_F;
      const float* nr = n + (size_t)rcv[edge] * GN_F;
      unsigned short* row = &lx[w][m][0];
      for (int c = lane; c < GN_F; c += 32) {
        row[c]            = f32_to_bf16(er[c]);
        row[GN_F + c]     = f32_to_bf16(ns[c]);
        row[2 * GN_F + c] = f32_to_bf16(nr[c]);
        row[3 * GN_F + c] = f32_to_bf16(g[c]);
      }
    }
  }
  __syncthreads();

  // ---- layer 1: [16x384] @ [384x96], bias preloaded into C
  v8f acc[GN_NT];
#pragma unroll
  for (int nt = 0; nt < GN_NT; ++nt) {
    const float bv = b1[nt * 16 + col];
#pragma unroll
    for (int r = 0; r < 8; ++r) acc[nt][r] = bv;
  }
  {
    const unsigned short* xrow = &lx[w][m16][0];
#pragma unroll
    for (int kt = 0; kt < 12; ++kt) {
      BFV a;
#pragma unroll
      for (int j = 0; j < 16; ++j)                       // A 16-bit layout
        a.u[j] = xrow[kt * 32 + gq * 8 + j + ((j >> 3) << 3)];
#pragma unroll
      for (int nt = 0; nt < GN_NT; ++nt) {
        BFV bb;
        bb.v = *(const v16bf*)(W1p + (((size_t)(kt * GN_NT + nt) * 32 + lane) << 4));
        acc[nt] = __builtin_amdgcn_wmma_f32_16x16x32_bf16(
            false, a.v, false, bb.v, (short)0, acc[nt], false, false);
      }
    }
  }
  __syncthreads();

  // ---- leaky relu, stage h (16x96 bf16) reusing the x LDS region
  unsigned short* hb = &lx[w][0][0];
#pragma unroll
  for (int nt = 0; nt < GN_NT; ++nt) {
#pragma unroll
    for (int r = 0; r < 8; ++r) {
      float v = acc[nt][r];
      v = (v > 0.0f) ? v : 0.01f * v;
      hb[(r + 8 * gq) * GN_F + nt * 16 + col] = f32_to_bf16(v);
    }
  }
  __syncthreads();

  // ---- layer 2: [16x96] @ [96x96]
  v8f acc2[GN_NT];
#pragma unroll
  for (int nt = 0; nt < GN_NT; ++nt) {
    const float bv = b2[nt * 16 + col];
#pragma unroll
    for (int r = 0; r < 8; ++r) acc2[nt][r] = bv;
  }
  {
    const unsigned short* hrow = hb + (size_t)m16 * GN_F;
#pragma unroll
    for (int kt = 0; kt < 3; ++kt) {
      BFV a;
#pragma unroll
      for (int j = 0; j < 16; ++j)
        a.u[j] = hrow[kt * 32 + gq * 8 + j + ((j >> 3) << 3)];
#pragma unroll
      for (int nt = 0; nt < GN_NT; ++nt) {
        BFV bb;
        bb.v = *(const v16bf*)(W2p + (((size_t)(kt * GN_NT + nt) * 32 + lane) << 4));
        acc2[nt] = __builtin_amdgcn_wmma_f32_16x16x32_bf16(
            false, a.v, false, bb.v, (short)0, acc2[nt], false, false);
      }
    }
  }

  // ---- epilogue: in-place e write + segment-sum scatter (D: M=r+8g, N=lane&15)
  if (active) {
#pragma unroll
    for (int nt = 0; nt < GN_NT; ++nt) {
      const int feat = nt * 16 + col;
#pragma unroll
      for (int r = 0; r < 8; ++r) {
        const int edge = e0 + r + 8 * gq;
        if (edge >= E) continue;
        const float v = acc2[nt][r];
        e[(size_t)edge * GN_F + feat] = v;
        atomicAdd(&sent[(size_t)snd[edge] * GN_F + feat], v);
        atomicAdd(&recvb[(size_t)rcv[edge] * GN_F + feat], v);
        atomicAdd(&esum[feat], v);
      }
    }
  }
}

// ---------------------------------------------------------------------------
// Fused node MLP step: x = [n | sent | recv | g]; in-place n write + nsum.
// ---------------------------------------------------------------------------
__global__ __launch_bounds__(32 * GN_WAVES) void node_update_kernel(
    float* __restrict__ n, const float* __restrict__ sent, const float* __restrict__ recvb,
    const float* __restrict__ g,
    const unsigned short* __restrict__ W1p, const float* __restrict__ b1,
    const unsigned short* __restrict__ W2p, const float* __restrict__ b2,
    float* __restrict__ nsum, int ntiles, int N)
{
  __shared__ unsigned short lx[GN_WAVES][16][4 * GN_F];
  const int lane = threadIdx.x & 31;
  const int w    = threadIdx.x >> 5;
  const int tile = blockIdx.x * GN_WAVES + w;
  const bool active = tile < ntiles;
  const int gq  = lane >> 4;
  const int col = lane & 15;
  const int m16 = lane & 15;
  const int n0  = tile * 16;

  if (active) {
    for (int m = 0; m < 16; ++m) {
      const int node = n0 + m;
      if (node >= N) break;
      const float* xn = n     + (size_t)node * GN_F;
      const float* xs = sent  + (size_t)node * GN_F;
      const float* xr = recvb + (size_t)node * GN_F;
      unsigned short* row = &lx[w][m][0];
      for (int c = lane; c < GN_F; c += 32) {
        row[c]            = f32_to_bf16(xn[c]);
        row[GN_F + c]     = f32_to_bf16(xs[c]);
        row[2 * GN_F + c] = f32_to_bf16(xr[c]);
        row[3 * GN_F + c] = f32_to_bf16(g[c]);
      }
    }
  }
  __syncthreads();

  v8f acc[GN_NT];
#pragma unroll
  for (int nt = 0; nt < GN_NT; ++nt) {
    const float bv = b1[nt * 16 + col];
#pragma unroll
    for (int r = 0; r < 8; ++r) acc[nt][r] = bv;
  }
  {
    const unsigned short* xrow = &lx[w][m16][0];
#pragma unroll
    for (int kt = 0; kt < 12; ++kt) {
      BFV a;
#pragma unroll
      for (int j = 0; j < 16; ++j)
        a.u[j] = xrow[kt * 32 + gq * 8 + j + ((j >> 3) << 3)];
#pragma unroll
      for (int nt = 0; nt < GN_NT; ++nt) {
        BFV bb;
        bb.v = *(const v16bf*)(W1p + (((size_t)(kt * GN_NT + nt) * 32 + lane) << 4));
        acc[nt] = __builtin_amdgcn_wmma_f32_16x16x32_bf16(
            false, a.v, false, bb.v, (short)0, acc[nt], false, false);
      }
    }
  }
  __syncthreads();

  unsigned short* hb = &lx[w][0][0];
#pragma unroll
  for (int nt = 0; nt < GN_NT; ++nt) {
#pragma unroll
    for (int r = 0; r < 8; ++r) {
      float v = acc[nt][r];
      v = (v > 0.0f) ? v : 0.01f * v;
      hb[(r + 8 * gq) * GN_F + nt * 16 + col] = f32_to_bf16(v);
    }
  }
  __syncthreads();

  v8f acc2[GN_NT];
#pragma unroll
  for (int nt = 0; nt < GN_NT; ++nt) {
    const float bv = b2[nt * 16 + col];
#pragma unroll
    for (int r = 0; r < 8; ++r) acc2[nt][r] = bv;
  }
  {
    const unsigned short* hrow = hb + (size_t)m16 * GN_F;
#pragma unroll
    for (int kt = 0; kt < 3; ++kt) {
      BFV a;
#pragma unroll
      for (int j = 0; j < 16; ++j)
        a.u[j] = hrow[kt * 32 + gq * 8 + j + ((j >> 3) << 3)];
#pragma unroll
      for (int nt = 0; nt < GN_NT; ++nt) {
        BFV bb;
        bb.v = *(const v16bf*)(W2p + (((size_t)(kt * GN_NT + nt) * 32 + lane) << 4));
        acc2[nt] = __builtin_amdgcn_wmma_f32_16x16x32_bf16(
            false, a.v, false, bb.v, (short)0, acc2[nt], false, false);
      }
    }
  }

  if (active) {
#pragma unroll
    for (int nt = 0; nt < GN_NT; ++nt) {
      const int feat = nt * 16 + col;
#pragma unroll
      for (int r = 0; r < 8; ++r) {
        const int node = n0 + r + 8 * gq;
        if (node >= N) continue;
        const float v = acc2[nt][r];
        n[(size_t)node * GN_F + feat] = v;
        atomicAdd(&nsum[feat], v);
      }
    }
  }
}

// ---------------------------------------------------------------------------
// Global update (tiny): g' = leaky([nsum|esum|g] @ gw1 + gb1) @ gw2 + gb2
// ---------------------------------------------------------------------------
__global__ void global_update_kernel(const float* __restrict__ nsum, const float* __restrict__ esum,
                                     float* __restrict__ g,
                                     const float* __restrict__ w1, const float* __restrict__ b1,
                                     const float* __restrict__ w2, const float* __restrict__ b2) {
  __shared__ float x[3 * GN_F];
  __shared__ float h[GN_F];
  const int t = threadIdx.x;
  x[t]            = nsum[t];
  x[GN_F + t]     = esum[t];
  x[2 * GN_F + t] = g[t];
  __syncthreads();
  float a = b1[t];
  for (int k = 0; k < 3 * GN_F; ++k) a += x[k] * w1[(size_t)k * GN_F + t];
  h[t] = (a > 0.0f) ? a : 0.01f * a;
  __syncthreads();
  float o = b2[t];
  for (int k = 0; k < GN_F; ++k) o += h[k] * w2[(size_t)k * GN_F + t];
  g[t] = o;
}

// ---------------------------------------------------------------------------
extern "C" void kernel_launch(void* const* d_in, const int* in_sizes, int n_in,
                              void* d_out, int out_size, void* d_ws, size_t ws_size,
                              hipStream_t stream) {
  (void)n_in; (void)out_size; (void)ws_size;
  const float* nodes    = (const float*)d_in[0];
  const float* edges    = (const float*)d_in[1];
  const float* globals_ = (const float*)d_in[2];
  const int*   senders  = (const int*)d_in[3];
  const int*   receivers= (const int*)d_in[4];
  const float* Wn  = (const float*)d_in[5];
  const float* bn  = (const float*)d_in[6];
  const float* We  = (const float*)d_in[7];
  const float* be  = (const float*)d_in[8];
  const float* Wg  = (const float*)d_in[9];
  const float* bg  = (const float*)d_in[10];
  const float* ew1 = (const float*)d_in[11];
  const float* eb1 = (const float*)d_in[12];
  const float* ew2 = (const float*)d_in[13];
  const float* eb2 = (const float*)d_in[14];
  const float* nw1 = (const float*)d_in[15];
  const float* nb1 = (const float*)d_in[16];
  const float* nw2 = (const float*)d_in[17];
  const float* nb2 = (const float*)d_in[18];
  const float* gw1 = (const float*)d_in[19];
  const float* gb1 = (const float*)d_in[20];
  const float* gw2 = (const float*)d_in[21];
  const float* gb2 = (const float*)d_in[22];

  const int N  = in_sizes[0] / 32;          // 50000
  const int E  = in_sizes[3];               // 800000
  const int S  = in_sizes[12] / GN_F;       // 3 steps
  const int K1 = 4 * GN_F;                  // 384
  const int K2 = GN_F;                      // 96

  // n, e, g live in d_out and are updated in place (row-local updates only).
  float* n = (float*)d_out;
  float* e = n + (size_t)N * GN_F;
  float* g = e + (size_t)E * GN_F;

  // workspace: segment sums + reduction vectors + packed bf16 weights
  float* sent  = (float*)d_ws;
  float* recvb = sent + (size_t)N * GN_F;
  float* nsum  = recvb + (size_t)N * GN_F;
  float* esum  = nsum + GN_F;
  unsigned short* ew1p = (unsigned short*)(esum + GN_F);
  unsigned short* ew2p = ew1p + (size_t)S * K1 * GN_F;
  unsigned short* nw1p = ew2p + (size_t)S * K2 * GN_F;
  unsigned short* nw2p = nw1p + (size_t)S * K1 * GN_F;

  // 1) pack weights into WMMA B-fragment bf16 layout (cheap, once per launch)
  {
    const int t1 = S * K1 * GN_F, t2 = S * K2 * GN_F;
    pack_b_kernel<<<(t1 + 255) / 256, 256, 0, stream>>>(ew1, ew1p, K1, t1);
    pack_b_kernel<<<(t2 + 255) / 256, 256, 0, stream>>>(ew2, ew2p, K2, t2);
    pack_b_kernel<<<(t1 + 255) / 256, 256, 0, stream>>>(nw1, nw1p, K1, t1);
    pack_b_kernel<<<(t2 + 255) / 256, 256, 0, stream>>>(nw2, nw2p, K2, t2);
  }
  // 2) embedders
  {
    const long long tn = (long long)N * GN_F;
    const long long te = (long long)E * GN_F;
    embed_kernel<<<(unsigned)((tn + 255) / 256), 256, 0, stream>>>(nodes, Wn, bn, n, N, 32);
    embed_kernel<<<(unsigned)((te + 255) / 256), 256, 0, stream>>>(edges, We, be, e, E, 16);
    embed_kernel<<<1, GN_F, 0, stream>>>(globals_, Wg, bg, g, 1, 8);
  }
  // 3) message-passing steps
  const int tilesE = (E + 15) / 16;
  const int tilesN = (N + 15) / 16;
  for (int s = 0; s < S; ++s) {
    const long long zn = 2LL * N * GN_F + 2 * GN_F;   // sent, recv, nsum, esum
    zero_kernel<<<(unsigned)((zn + 255) / 256), 256, 0, stream>>>(sent, zn);

    edge_update_kernel<<<(tilesE + GN_WAVES - 1) / GN_WAVES, 32 * GN_WAVES, 0, stream>>>(
        e, n, g, senders, receivers,
        ew1p + (size_t)s * K1 * GN_F, eb1 + s * GN_F,
        ew2p + (size_t)s * K2 * GN_F, eb2 + s * GN_F,
        sent, recvb, esum, tilesE, E);

    node_update_kernel<<<(tilesN + GN_WAVES - 1) / GN_WAVES, 32 * GN_WAVES, 0, stream>>>(
        n, sent, recvb, g,
        nw1p + (size_t)s * K1 * GN_F, nb1 + s * GN_F,
        nw2p + (size_t)s * K2 * GN_F, nb2 + s * GN_F,
        nsum, tilesN, N);

    global_update_kernel<<<1, GN_F, 0, stream>>>(
        nsum, esum, g,
        gw1 + (size_t)s * 3 * GN_F * GN_F, gb1 + s * GN_F,
        gw2 + (size_t)s * GN_F * GN_F,     gb2 + s * GN_F);
  }
}
